// PatchNCELoss_7447473291936
// MI455X (gfx1250) — matrix operand order repaired and checked by reading, not compiled, not run on hardware
//
#include <hip/hip_runtime.h>
#include <hip/hip_bf16.h>
#include <stdint.h>

// Problem constants from the reference
#define BN 4
#define CN 256
#define SN 4096            // H*W = 64*64
#define NTOT ((size_t)BN * SN * CN)   // elements per split array
#define NWG  (BN * (SN / 16))         // 1024 row-tile workgroups

typedef __attribute__((ext_vector_type(16))) __bf16 v16bf;
typedef __attribute__((ext_vector_type(8)))  float  v8f;

static __device__ inline v16bf mk16(uint4 a, uint4 b) {
    union { v16bf v; uint4 u[2]; } t;
    t.u[0] = a; t.u[1] = b;
    return t.v;
}

// --- Kernel 1: transpose [C,S] -> [S,C] and split fp32 into bf16 hi/lo ---
__global__ __launch_bounds__(256) void convert_split(
    const float* __restrict__ fq, const float* __restrict__ fk,
    uint16_t* __restrict__ qh, uint16_t* __restrict__ ql,
    uint16_t* __restrict__ kh, uint16_t* __restrict__ kl)
{
    const int isK = blockIdx.z & 1;
    const int b   = blockIdx.z >> 1;
    const float* src = isK ? fk : fq;
    uint16_t* oh = isK ? kh : qh;
    uint16_t* ol = isK ? kl : ql;
    const int s0 = blockIdx.x * 32;
    const int c0 = blockIdx.y * 32;

    __shared__ float tile[32][33];
    #pragma unroll
    for (int i = 0; i < 4; ++i) {
        int cr = threadIdx.y + i * 8;
        tile[cr][threadIdx.x] = src[((size_t)b * CN + (c0 + cr)) * SN + s0 + threadIdx.x];
    }
    __syncthreads();
    #pragma unroll
    for (int i = 0; i < 4; ++i) {
        int sr = threadIdx.y + i * 8;
        float v = tile[threadIdx.x][sr];
        uint32_t u  = __float_as_uint(v);
        uint32_t hb = (u + 0x7FFFu + ((u >> 16) & 1u)) >> 16;       // RNE bf16 hi
        float hf = __uint_as_float(hb << 16);
        float lv = v - hf;                                           // residual
        uint32_t u2 = __float_as_uint(lv);
        uint32_t lb = (u2 + 0x7FFFu + ((u2 >> 16) & 1u)) >> 16;     // RNE bf16 lo
        size_t o = ((size_t)b * SN + (s0 + sr)) * CN + c0 + threadIdx.x;
        oh[o] = (uint16_t)hb;
        ol[o] = (uint16_t)lb;
    }
}

// --- Kernel 2: fused gram-GEMM (split-bf16 WMMA) + online row LSE ---
#define AROW 33   // uint4 per LDS A row (32 data + 1 pad -> no bank conflicts)

__global__ __launch_bounds__(256) void nce_main(
    const uint16_t* __restrict__ qh, const uint16_t* __restrict__ ql,
    const uint16_t* __restrict__ kh, const uint16_t* __restrict__ kl,
    float* __restrict__ partials)
{
    __shared__ uint4  ldsAh[16 * AROW];
    __shared__ uint4  ldsAl[16 * AROW];
    __shared__ float2 wst[8][16];
    __shared__ float  dg[16];
    __shared__ float  lrow[16];

    const int wg = blockIdx.x;
    const int b  = wg >> 8;          // 256 row tiles per batch
    const int rt = wg & 255;
    const int row_base = rt * 16;

    const int tid  = threadIdx.x;
    const int wave = tid >> 5;
    const int lane = tid & 31;
    const int lm   = lane & 15;
    const int half = lane >> 4;

    // Stage A tile (16 rows x 256 ch, hi+lo) into LDS; rows are contiguous 512B
    {
        const uint4* gqh = (const uint4*)(qh + ((size_t)b * SN + row_base) * CN);
        const uint4* gql = (const uint4*)(ql + ((size_t)b * SN + row_base) * CN);
        for (int i = tid; i < 512; i += 256) {
            int r = i >> 5, k = i & 31;
            ldsAh[r * AROW + k] = gqh[i];
            ldsAl[r * AROW + k] = gql[i];
        }
    }
    __syncthreads();

    const float kScale = 1.0f / (0.07f * 0.69314718056f);   // 1/(tau*ln2): work in log2 domain
    float rmax[8], rsum[8];
    #pragma unroll
    for (int r = 0; r < 8; ++r) { rmax[r] = -__builtin_inff(); rsum[r] = 0.0f; }

    const uint16_t* bhbase = kh + (size_t)b * SN * CN;
    const uint16_t* blbase = kl + (size_t)b * SN * CN;

    for (int ct = 0; ct < 32; ++ct) {
        const int j = ct * 8 + wave;           // global 16-column tile index
        const int col = j * 16 + lm;           // this lane's N column
        const uint4* bh = (const uint4*)(bhbase + (size_t)col * CN) + half;
        const uint4* bl = (const uint4*)(blbase + (size_t)col * CN) + half;
        if (ct + 1 < 32)
            __builtin_prefetch(bhbase + (size_t)(col + 128) * CN, 0, 0);

        v8f acc = {};
        #pragma unroll
        for (int kc = 0; kc < 8; ++kc) {
            const int a1 = lm * AROW + kc * 4 + half;      // 16B piece 1; piece 2 at +32B
            v16bf ah = mk16(ldsAh[a1], ldsAh[a1 + 2]);
            v16bf al = mk16(ldsAl[a1], ldsAl[a1 + 2]);
            const uint4* bhp = bh + kc * 4;                // 16B piece 1; piece 2 at +32B
            const uint4* blp = bl + kc * 4;
            v16bf bhv = mk16(bhp[0], bhp[2]);
            v16bf blv = mk16(blp[0], blp[2]);
            acc = __builtin_amdgcn_wmma_f32_16x16x32_bf16(false, al, false, bhv, (short)0, acc, false, false);
            acc = __builtin_amdgcn_wmma_f32_16x16x32_bf16(false, ah, false, blv, (short)0, acc, false, false);
            acc = __builtin_amdgcn_wmma_f32_16x16x32_bf16(false, ah, false, bhv, (short)0, acc, false, false);
        }

        // Diagonal tile: element (m,m) lives in acc[m] of lane m (m<8) and
        // acc[m-8] of lane m+16 (m>=8)
        if (j == rt) {
            float av[8];
            #pragma unroll
            for (int r = 0; r < 8; ++r) av[r] = acc[r];
            int r = -1, m = 0;
            if (lane < 8)        { r = lane;      m = lane; }
            else if (lane >= 24) { r = lane - 24; m = lane - 16; }
            if (r >= 0) {
                float x = 0.0f;
                #pragma unroll
                for (int i = 0; i < 8; ++i) if (i == r) x = av[i];
                dg[m] = x * kScale;
            }
        }

        // Online log2-domain softmax update (rows m = r + half*8)
        #pragma unroll
        for (int r = 0; r < 8; ++r) {
            float x  = acc[r] * kScale;
            float mo = rmax[r];
            float mn = fmaxf(mo, x);
            rsum[r]  = rsum[r] * __builtin_amdgcn_exp2f(mo - mn)
                     + __builtin_amdgcn_exp2f(x - mn);
            rmax[r]  = mn;
        }
    }

    // Reduce (max,sumexp2) across the 16 lanes of each half (xor 1,2,4,8)
    #pragma unroll
    for (int off = 1; off < 16; off <<= 1) {
        #pragma unroll
        for (int r = 0; r < 8; ++r) {
            float m2 = __shfl_xor(rmax[r], off, 32);
            float s2 = __shfl_xor(rsum[r], off, 32);
            float mn = fmaxf(rmax[r], m2);
            rsum[r] = rsum[r] * __builtin_amdgcn_exp2f(rmax[r] - mn)
                    + s2      * __builtin_amdgcn_exp2f(m2 - mn);
            rmax[r] = mn;
        }
    }
    if (lm == 0) {   // lanes 0 and 16 publish per-row state
        #pragma unroll
        for (int r = 0; r < 8; ++r)
            wst[wave][half * 8 + r] = make_float2(rmax[r], rsum[r]);
    }
    __syncthreads();

    // Merge across the 8 waves; per-row loss = ln2*(m + log2(s) - diag)
    if (tid < 16) {
        float m = -__builtin_inff(), s = 0.0f;
        #pragma unroll
        for (int w = 0; w < 8; ++w) {
            float2 p = wst[w][tid];
            float mn = fmaxf(m, p.x);
            s = s * __builtin_amdgcn_exp2f(m - mn)
              + p.y * __builtin_amdgcn_exp2f(p.x - mn);
            m = mn;
        }
        lrow[tid] = 0.69314718056f * (m + __builtin_amdgcn_logf(s) - dg[tid]);
    }
    __syncthreads();
    if (tid == 0) {
        float t = 0.0f;
        #pragma unroll
        for (int i = 0; i < 16; ++i) t += lrow[i];
        partials[wg] = t;
    }
}

// --- Kernel 3: deterministic reduction of per-workgroup partials ---
__global__ __launch_bounds__(256) void reduce_partials(
    const float* __restrict__ p, float* __restrict__ out)
{
    __shared__ float sm[256];
    float s = 0.0f;
    for (int i = threadIdx.x; i < NWG; i += 256) s += p[i];
    sm[threadIdx.x] = s;
    __syncthreads();
    for (int off = 128; off > 0; off >>= 1) {
        if (threadIdx.x < off) sm[threadIdx.x] += sm[threadIdx.x + off];
        __syncthreads();
    }
    if (threadIdx.x == 0) out[0] = sm[0] * (1.0f / (float)(BN * SN));
}

extern "C" void kernel_launch(void* const* d_in, const int* in_sizes, int n_in,
                              void* d_out, int out_size, void* d_ws, size_t ws_size,
                              hipStream_t stream)
{
    (void)in_sizes; (void)n_in; (void)out_size; (void)ws_size;
    const float* fq = (const float*)d_in[0];
    const float* fk = (const float*)d_in[1];

    uint16_t* qh = (uint16_t*)d_ws;
    uint16_t* ql = qh + NTOT;
    uint16_t* kh = ql + NTOT;
    uint16_t* kl = kh + NTOT;
    float* partials = (float*)(kl + NTOT);
    float* out = (float*)d_out;

    convert_split<<<dim3(SN / 32, CN / 32, BN * 2), dim3(32, 8), 0, stream>>>(
        fq, fk, qh, ql, kh, kl);
    nce_main<<<NWG, 256, 0, stream>>>(qh, ql, kh, kl, partials);
    reduce_partials<<<1, 256, 0, stream>>>(partials, out);
}
